// Attention_8400956031480
// MI455X (gfx1250) — compile-verified, matrix operand order checked by reading, not run on hardware
//
#include <hip/hip_runtime.h>
#include <hip/hip_bf16.h>

// CDNA5 (gfx1250) wave32 WMMA attention:
//  * proj_kernel<TRANSPOSE>: one wave computes a 16x64 output strip with 4
//    accumulators; A operand loaded once per K-step. Q is pre-scaled by
//    (1/sqrt(H)) * log2(e) so the softmax runs in the exp2 domain with no
//    per-element multiplies in the hot loop.
//  * attn_kernel: 4 waves / workgroup share double-buffered 64-key K and V^T
//    blocks in LDS via global_load_async_to_lds_b128 (ASYNCcnt); WMMA operands
//    come from ds_load_b128. Causal mask applied only on the diagonal block.
//    Softmax row reductions use DPP16 butterflies (no LDS round-trips).
//
// WMMA 16-bit A/B operand lane layout (wave32):
//   lane group g = lane>>4; vector element j in [0,16):
//     K = 8*g + j        for j in [0,8)
//     K = 8*g + 16 + j-8 for j in [8,16)
// f32 C/D layout: VGPR r holds row (r + 8*g), column = lane&15.

typedef __attribute__((ext_vector_type(16))) __bf16 v16bf;
typedef __attribute__((ext_vector_type(8)))  __bf16 v8bf;
typedef __attribute__((ext_vector_type(8)))  float  v8f;
typedef __attribute__((ext_vector_type(4)))  float  v4f;
typedef __attribute__((ext_vector_type(4)))  int    v4i;

typedef __attribute__((address_space(1))) v4i as1_v4i;   // global int4
typedef __attribute__((address_space(3))) v4i as3_v4i;   // LDS int4

#define B_ 4
#define S_ 4096
#define D_ 1024
#define H_ 64
#define KB 64   // key block per workgroup iteration

#if defined(__has_builtin)
#if __has_builtin(__builtin_amdgcn_global_load_async_to_lds_b128)
#define HAS_ASYNC 1
#endif
#if __has_builtin(__builtin_amdgcn_update_dpp)
#define HAS_DPP 1
#endif
#endif

__device__ __forceinline__ void copy16(const __bf16* g, __bf16* l) {
#ifdef HAS_ASYNC
  __builtin_amdgcn_global_load_async_to_lds_b128(
      (as1_v4i*)g, (as3_v4i*)l, 0, 0);
#else
  *(v8bf*)l = *(const v8bf*)g;
#endif
}
__device__ __forceinline__ void wait_async_le8() {
#ifdef HAS_ASYNC
  asm volatile("s_wait_asynccnt 0x8" ::: "memory");
#endif
}
__device__ __forceinline__ void wait_async_0() {
#ifdef HAS_ASYNC
  asm volatile("s_wait_asynccnt 0x0" ::: "memory");
#endif
}

// exp2 that maps directly to v_exp_f32 (TRANS op, co-executes with WMMA)
__device__ __forceinline__ float fast_exp2(float x) {
#if defined(__has_builtin) && __has_builtin(__builtin_amdgcn_exp2f)
  return __builtin_amdgcn_exp2f(x);
#else
  return __expf(x * 0.6931471805599453f);
#endif
}

// ---- DPP16 butterfly reductions across a 16-lane group (pure VALU) ----
#ifdef HAS_DPP
template <int CTRL>
__device__ __forceinline__ float dppmov(float x) {
  return __builtin_bit_cast(float, __builtin_amdgcn_update_dpp(
      0, __builtin_bit_cast(int, x), CTRL, 0xF, 0xF, true));
}
__device__ __forceinline__ float rowmax16(float x) {
  x = fmaxf(x, dppmov<0xB1>(x));    // quad_perm(1,0,3,2): xor 1
  x = fmaxf(x, dppmov<0x4E>(x));    // quad_perm(2,3,0,1): xor 2
  x = fmaxf(x, dppmov<0x141>(x));   // row_half_mirror:    xor 7
  x = fmaxf(x, dppmov<0x140>(x));   // row_mirror:         xor 15
  return x;
}
__device__ __forceinline__ float rowsum16(float x) {
  x += dppmov<0xB1>(x);
  x += dppmov<0x4E>(x);
  x += dppmov<0x141>(x);
  x += dppmov<0x140>(x);
  return x;
}
#else
__device__ __forceinline__ float rowmax16(float x) {
  for (int off = 1; off < 16; off <<= 1) x = fmaxf(x, __shfl_xor(x, off, 32));
  return x;
}
__device__ __forceinline__ float rowsum16(float x) {
  for (int off = 1; off < 16; off <<= 1) x += __shfl_xor(x, off, 32);
  return x;
}
#endif

// ---------------------------------------------------------------------------
// Projection: out = (x @ W^T) * scl, fp32 -> bf16. Optionally transposed store.
// One wave per 16-row strip, all 4 h-tiles (H=64), K-loop over D in steps of 32.
// ---------------------------------------------------------------------------
template <int TRANSPOSE>
__global__ __launch_bounds__(32)
void proj_kernel(const float* __restrict__ x, const float* __restrict__ W,
                 __bf16* __restrict__ out, float scl) {
  const int lane = threadIdx.x;
  const int g  = lane >> 4;
  const int nr = lane & 15;
  const int rowTile = blockIdx.x;                       // 0 .. B*S/16-1

  const float* xrow = x + (size_t)(rowTile * 16 + nr) * D_;   // A row = nr

  v8f acc[4] = {};
  for (int d0 = 0; d0 < D_; d0 += 32) {
    v4f a0 = *(const v4f*)(xrow + d0 + 8 * g);
    v4f a1 = *(const v4f*)(xrow + d0 + 8 * g + 4);
    v4f a2 = *(const v4f*)(xrow + d0 + 8 * g + 16);
    v4f a3 = *(const v4f*)(xrow + d0 + 8 * g + 20);
    v16bf av;
#pragma unroll
    for (int j = 0; j < 4; ++j) {
      av[j]      = (__bf16)a0[j];
      av[j + 4]  = (__bf16)a1[j];
      av[j + 8]  = (__bf16)a2[j];
      av[j + 12] = (__bf16)a3[j];
    }
#pragma unroll
    for (int ht = 0; ht < 4; ++ht) {                    // B col = ht*16 + nr
      const float* wrow = W + (size_t)(ht * 16 + nr) * D_;
      v4f b0 = *(const v4f*)(wrow + d0 + 8 * g);
      v4f b1 = *(const v4f*)(wrow + d0 + 8 * g + 4);
      v4f b2 = *(const v4f*)(wrow + d0 + 8 * g + 16);
      v4f b3 = *(const v4f*)(wrow + d0 + 8 * g + 20);
      v16bf bv;
#pragma unroll
      for (int j = 0; j < 4; ++j) {
        bv[j]      = (__bf16)b0[j];
        bv[j + 4]  = (__bf16)b1[j];
        bv[j + 8]  = (__bf16)b2[j];
        bv[j + 12] = (__bf16)b3[j];
      }
      acc[ht] = __builtin_amdgcn_wmma_f32_16x16x32_bf16(false, av, false, bv,
                                                        (short)0, acc[ht],
                                                        false, false);
    }
  }
  // C layout: VGPR r -> flattened row m = tile*16 + r + 8*g, col = ht*16 + nr
#pragma unroll
  for (int ht = 0; ht < 4; ++ht) {
#pragma unroll
    for (int r = 0; r < 8; ++r) {
      int m = rowTile * 16 + r + 8 * g;
      int col = ht * 16 + nr;
      float val = acc[ht][r] * scl;
      if (TRANSPOSE) {                    // V^T: [B][H][S], s contiguous
        int b = m >> 12;                  // / S_
        int s = m & (S_ - 1);
        out[((size_t)b * H_ + col) * S_ + s] = (__bf16)val;
      } else {
        out[(size_t)m * H_ + col] = (__bf16)val;
      }
    }
  }
}

// ---------------------------------------------------------------------------
// Flash attention: 128 threads (4 waves) per WG; wave w owns query rows
// [q0+16w, q0+16w+16). K and V^T key-blocks of 64 are double-buffered in LDS.
// ---------------------------------------------------------------------------
__global__ __launch_bounds__(128)
void attn_kernel(const __bf16* __restrict__ Q, const __bf16* __restrict__ K,
                 const __bf16* __restrict__ Vt, float* __restrict__ out) {
  __shared__ __align__(16) __bf16 Klds[2][KB][H_];   // 16 KB
  __shared__ __align__(16) __bf16 Vlds[2][H_][KB];   // 16 KB
  __shared__ __align__(16) __bf16 Plds[4][16][KB];   // 8 KB (per-wave slices)

  const int tid  = threadIdx.x;
  const int wave = tid >> 5;
  const int lane = tid & 31;
  const int g  = lane >> 4;
  const int nr = lane & 15;

  const int bq = blockIdx.x & (S_ / KB - 1);          // q-block within batch
  const int b  = blockIdx.x / (S_ / KB);
  const int q0 = bq * KB;
  const int s0w = q0 + wave * 16;                     // this wave's first q row
  const int nblk = bq + 1;                            // causal: keys < q0+64

  // cooperative K/V staging: 8 x 16B per thread per block (8 async ops / wave)
  auto stage = [&](int blk, int buf) {
    const __bf16* ksrc = K + ((size_t)b * S_ + blk * KB) * H_;  // 8KB contiguous
    __bf16* kdst = &Klds[buf][0][0];
#pragma unroll
    for (int i = 0; i < 4; ++i) {
      int e = (tid + i * 128) * 8;
      copy16(ksrc + e, kdst + e);
    }
#pragma unroll
    for (int i = 0; i < 4; ++i) {
      int c  = tid + i * 128;                         // 512 chunks of 16B
      int h  = c >> 3;
      int sl = (c & 7) * 8;
      copy16(Vt + ((size_t)b * H_ + h) * S_ + blk * KB + sl, &Vlds[buf][h][sl]);
    }
  };

  // Q tile (16 q x 64 h) as two 16x32 bf16 A-operands, in registers
  v16bf qa[2];
  {
    const __bf16* qrow = Q + ((size_t)b * S_ + s0w + nr) * H_;
#pragma unroll
    for (int hs = 0; hs < 2; ++hs) {
      v8bf lo = *(const v8bf*)(qrow + hs * 32 + 8 * g);
      v8bf hi = *(const v8bf*)(qrow + hs * 32 + 8 * g + 16);
#pragma unroll
      for (int j = 0; j < 8; ++j) { qa[hs][j] = lo[j]; qa[hs][j + 8] = hi[j]; }
    }
  }

  v8f o[4] = {};                    // 16 q x 64 h fp32 accumulators
  float mrow[8], lrow[8];           // running max / sum in log2 domain
#pragma unroll
  for (int r = 0; r < 8; ++r) { mrow[r] = -1e30f; lrow[r] = 0.f; }

  // One key block: scores -> online softmax -> P@V.  MASK only on the
  // diagonal block (i == nblk-1); earlier blocks are fully unmasked.
  auto process = [&](int i, bool MASK) {
    const int cur = i & 1;
    const int kb = i * KB;
    // ---- scores: hs outer, 4 independent accumulator chains inner ----
    v8f sc[4] = {};
#pragma unroll
    for (int hs = 0; hs < 2; ++hs) {
#pragma unroll
      for (int t = 0; t < 4; ++t) {
        v8bf lo = *(const v8bf*)(&Klds[cur][t * 16 + nr][hs * 32 + 8 * g]);
        v8bf hi = *(const v8bf*)(&Klds[cur][t * 16 + nr][hs * 32 + 8 * g + 16]);
        v16bf bv;
#pragma unroll
        for (int j = 0; j < 8; ++j) { bv[j] = lo[j]; bv[j + 8] = hi[j]; }
        sc[t] = __builtin_amdgcn_wmma_f32_16x16x32_bf16(false, qa[hs], false, bv,
                                                        (short)0, sc[t],
                                                        false, false);
      }
    }
    // ---- online softmax in exp2 domain (Q pre-scaled by log2e/sqrt(H)) ----
#pragma unroll
    for (int r = 0; r < 8; ++r) {
      int qi = s0w + r + 8 * g;
      float v[4];
#pragma unroll
      for (int t = 0; t < 4; ++t) {
        v[t] = sc[t][r];
        if (MASK && (kb + t * 16 + nr > qi)) v[t] = -1e30f;
      }
      float rm = rowmax16(fmaxf(fmaxf(v[0], v[1]), fmaxf(v[2], v[3])));
      float mnew = fmaxf(mrow[r], rm);
      float fac  = fast_exp2(mrow[r] - mnew);
      float p[4], rs = 0.f;
#pragma unroll
      for (int t = 0; t < 4; ++t) { p[t] = fast_exp2(v[t] - mnew); rs += p[t]; }
      rs = rowsum16(rs);
      lrow[r] = lrow[r] * fac + rs;
      mrow[r] = mnew;
#pragma unroll
      for (int ht = 0; ht < 4; ++ht) o[ht][r] *= fac;
      int prow = r + 8 * g;
#pragma unroll
      for (int t = 0; t < 4; ++t)
        Plds[wave][prow][t * 16 + nr] = (__bf16)p[t];
    }
    // ---- P as two 16x32 A-operands (own wave's slice; DS in-order) ----
    v16bf pa[2];
#pragma unroll
    for (int ks = 0; ks < 2; ++ks) {
      v8bf lo = *(const v8bf*)(&Plds[wave][nr][ks * 32 + 8 * g]);
      v8bf hi = *(const v8bf*)(&Plds[wave][nr][ks * 32 + 8 * g + 16]);
#pragma unroll
      for (int j = 0; j < 8; ++j) { pa[ks][j] = lo[j]; pa[ks][j + 8] = hi[j]; }
    }
    // ---- P @ V: ks outer, 4 independent accumulator chains inner ----
#pragma unroll
    for (int ks = 0; ks < 2; ++ks) {
#pragma unroll
      for (int ht = 0; ht < 4; ++ht) {
        v8bf lo = *(const v8bf*)(&Vlds[cur][ht * 16 + nr][ks * 32 + 8 * g]);
        v8bf hi = *(const v8bf*)(&Vlds[cur][ht * 16 + nr][ks * 32 + 8 * g + 16]);
        v16bf bv;
#pragma unroll
        for (int j = 0; j < 8; ++j) { bv[j] = lo[j]; bv[j + 8] = hi[j]; }
        o[ht] = __builtin_amdgcn_wmma_f32_16x16x32_bf16(false, pa[ks], false, bv,
                                                        (short)0, o[ht],
                                                        false, false);
      }
    }
  };

  stage(0, 0);
  for (int i = 0; i < nblk; ++i) {
    const int cur = i & 1;
    if (i + 1 < nblk) { stage(i + 1, 1 - cur); wait_async_le8(); }
    else              { wait_async_0(); }
    __syncthreads();                // buf[cur] fully written by all waves

    if (i + 1 < nblk) process(i, false);   // interior: no causal mask
    else             process(i, true);     // diagonal block

    __syncthreads();                // everyone done reading buf[cur]
  }

  // ---- normalize and store fp32 output [B][S][H] ----
#pragma unroll
  for (int r = 0; r < 8; ++r) {
    float inv = 1.0f / lrow[r];
    size_t row = (size_t)b * S_ + s0w + r + 8 * g;
#pragma unroll
    for (int ht = 0; ht < 4; ++ht)
      out[row * H_ + ht * 16 + nr] = o[ht][r] * inv;
  }
}

extern "C" void kernel_launch(void* const* d_in, const int* in_sizes, int n_in,
                              void* d_out, int out_size, void* d_ws, size_t ws_size,
                              hipStream_t stream) {
  const float* q  = (const float*)d_in[0];
  const float* k  = (const float*)d_in[1];
  const float* v  = (const float*)d_in[2];
  const float* Wq = (const float*)d_in[3];
  const float* Wk = (const float*)d_in[4];
  const float* Wv = (const float*)d_in[5];

  __bf16* Qbf = (__bf16*)d_ws;                       // [B][S][H] bf16, 2 MB
  __bf16* Kbf = Qbf + (size_t)B_ * S_ * H_;          // [B][S][H] bf16, 2 MB
  __bf16* Vt  = Kbf + (size_t)B_ * S_ * H_;          // [B][H][S] bf16, 2 MB

  // Q scale = (1/sqrt(H)) * log2(e): softmax runs in the exp2 domain.
  const float qscale = 0.125f * 1.44269504088896340736f;

  dim3 pg(B_ * S_ / 16);
  proj_kernel<0><<<pg, 32, 0, stream>>>(q, Wq, Qbf, qscale);
  proj_kernel<0><<<pg, 32, 0, stream>>>(k, Wk, Kbf, 1.0f);
  proj_kernel<1><<<pg, 32, 0, stream>>>(v, Wv, Vt, 1.0f);

  attn_kernel<<<dim3(B_ * (S_ / KB)), 128, 0, stream>>>(Qbf, Kbf, Vt,
                                                        (float*)d_out);
}
